// SegmentationHead_2439541424196
// MI455X (gfx1250) — matrix operand order repaired and checked by reading, not compiled, not run on hardware
//
#include <hip/hip_runtime.h>
#include <math.h>

// ---------------------------------------------------------------------------
// Problem constants (match reference)
// ---------------------------------------------------------------------------
namespace {
constexpr int Bz   = 4;
constexpr int Nn   = 16384;
constexpr int Ss   = 512;
constexpr int Pp   = 32;
constexpr int Tt   = Pp + Ss;        // 544
constexpr int INC  = 384;
constexpr int HID  = 256;
constexpr int MLP  = 4 * HID;        // 1024
constexpr int OUTC = 50;
constexpr int HEADS= 8;
constexpr int HD   = HID / HEADS;    // 32
constexpr int EMB  = 64;
constexpr int OBJD = 64;
constexpr int Gg   = 128;
constexpr int MT   = Bz * Tt;        // 2176 tokens (multiple of 32)
constexpr long MP  = (long)Bz * Nn;  // 65536 points (multiple of 32)
constexpr int FPK  = 800;            // 771 padded to multiple of 32
constexpr int CLSK = 896;            // 512+256+64+64, already mult of 32
}

typedef _Float16 v16h __attribute__((ext_vector_type(16)));
typedef _Float16 v8h  __attribute__((ext_vector_type(8)));
typedef float    v8f  __attribute__((ext_vector_type(8)));

// ---------------------------------------------------------------------------
// Elementwise f32 -> f16
// ---------------------------------------------------------------------------
__global__ __launch_bounds__(256)
void k_f32_to_f16(const float* __restrict__ src, _Float16* __restrict__ dst, long n) {
    long i = (long)blockIdx.x * 256 + threadIdx.x;
    if (i < n) dst[i] = (_Float16)src[i];
}

// ---------------------------------------------------------------------------
// Pack a row-major f32 weight W[K][Nc] into WMMA B-fragment order (f16).
// Tile t = (kb/32)*tilesN + tn holds 512 halves; within a tile, lane l owns
// halves [l*16 .. l*16+15] with b[i] = W[kb + (i>>3)*16 + (l>>4)*8 + (i&7)][tn*16 + (l&15)].
// Zero-pads K->Kp and Nc->Np.
// ---------------------------------------------------------------------------
__global__ __launch_bounds__(256)
void k_pack_b(const float* __restrict__ W, int K, int Nc, int Kp, int Np,
              _Float16* __restrict__ out, long total) {
    long id = (long)blockIdx.x * 256 + threadIdx.x;
    if (id >= total) return;
    int  i     = (int)(id & 15);
    int  lane  = (int)((id >> 4) & 31);
    long t     = id >> 9;
    int  tilesN= Np >> 4;
    int  kb    = (int)(t / tilesN) * 32;
    int  tn    = (int)(t % tilesN);
    int  n     = tn * 16 + (lane & 15);
    int  k     = kb + ((i >> 3) << 4) + ((lane >> 4) << 3) + (i & 7);
    float v = (k < K && n < Nc) ? W[(size_t)k * Nc + n] : 0.0f;
    out[id] = (_Float16)v;
}

// ---------------------------------------------------------------------------
// Fold linear bias + BatchNorm into per-channel scale/shift:
//   y' = y*scale + shift ; scale = g*rsqrt(v+eps) ; shift = (b - m)*scale + beta
// ---------------------------------------------------------------------------
__global__ __launch_bounds__(256)
void k_bn_fold(const float* __restrict__ lb, const float* __restrict__ g,
               const float* __restrict__ beta, const float* __restrict__ m,
               const float* __restrict__ var, float* __restrict__ scale,
               float* __restrict__ shift, int n) {
    int i = blockIdx.x * 256 + threadIdx.x;
    if (i >= n) return;
    float s = g[i] * rsqrtf(var[i] + 1e-5f);
    scale[i] = s;
    shift[i] = (lb[i] - m[i]) * s + beta[i];
}

// ---------------------------------------------------------------------------
// WMMA GEMM: Y = act(affine(A @ B) + resid)
//   A: f16 row-major [M x Kp] (lda), Kp mult of 32, M mult of 32
//   Bp: packed f16 B fragments (see k_pack_b), N mult of 64
//   One wave per 32x64 register tile: 2 A frags x 4 B frags -> 8 WMMAs per
//   K-step from 6 KB of fragment loads (0.75 KB / wmma of L2 traffic).
//   act: 0 none, 1 relu, 2 exact gelu. If scale!=null: y = y*scale+shift,
//   else y += bias (if bias). Optional residual add (f16). Optional f32 out.
// ---------------------------------------------------------------------------
__global__ __launch_bounds__(256)
void k_wmma_gemm(const _Float16* __restrict__ A, int lda,
                 const _Float16* __restrict__ Bp,
                 const float* __restrict__ bias,
                 const float* __restrict__ scale,
                 const float* __restrict__ shift,
                 const _Float16* __restrict__ resid, int ldr,
                 _Float16* __restrict__ outh, int ldo,
                 float* __restrict__ outf, int ldof, int nstore,
                 int M, int N, int Kp, int act) {
    const int lane = threadIdx.x & 31;
    const int wave = threadIdx.x >> 5;
    const int tn64 = N >> 6;                 // 64-wide N supertiles
    const int tm32 = M >> 5;                 // 32-tall M supertiles
    int gid = blockIdx.x * 8 + wave;
    int tile_m = gid / tn64;
    int tile_n = gid - tile_m * tn64;
    if (tile_m >= tm32) return;              // wave-uniform: EXEC stays all-ones

    const int khalf = lane >> 4;             // which K half this lane holds
    const int nl    = lane & 15;

    const v8f vzero = {0.f, 0.f, 0.f, 0.f, 0.f, 0.f, 0.f, 0.f};
    v8f acc[2][4];
#pragma unroll
    for (int i = 0; i < 2; ++i)
#pragma unroll
        for (int j = 0; j < 4; ++j) acc[i][j] = vzero;

    const _Float16* aptr = A + (size_t)(tile_m * 32 + nl) * lda + khalf * 8;
    const size_t arow16 = (size_t)16 * lda;  // second 16-row A fragment
    const _Float16* bptr = Bp + (size_t)(tile_n * 4) * 512 + (size_t)lane * 16;
    const int tilesN16 = N >> 4;
    const size_t bstride = (size_t)tilesN16 * 512;

    for (int kb = 0; kb < Kp; kb += 32) {
        union { v16h v; v8h h[2]; } a0, a1;
        a0.h[0] = *(const v8h*)(aptr);               // K = kb + khalf*8 + [0..7]
        a0.h[1] = *(const v8h*)(aptr + 16);          // K = kb + 16 + khalf*8 + [0..7]
        a1.h[0] = *(const v8h*)(aptr + arow16);
        a1.h[1] = *(const v8h*)(aptr + arow16 + 16);
#pragma unroll
        for (int j = 0; j < 4; ++j) {
            v16h b = *(const v16h*)(bptr + (size_t)j * 512);
            acc[0][j] = __builtin_amdgcn_wmma_f32_16x16x32_f16(
                            false, a0.v, false, b, (short)0, acc[0][j], false, false);
            acc[1][j] = __builtin_amdgcn_wmma_f32_16x16x32_f16(
                            false, a1.v, false, b, (short)0, acc[1][j], false, false);
        }
        aptr += 32;
        bptr += bstride;
    }

#pragma unroll
    for (int i = 0; i < 2; ++i) {
        const int m0 = tile_m * 32 + i * 16 + khalf * 8;
#pragma unroll
        for (int j = 0; j < 4; ++j) {
            const int col = tile_n * 64 + j * 16 + nl;
            const bool valid = col < nstore;
            float sc = (scale && valid) ? scale[col] : 1.0f;
            float sh = (scale && valid) ? shift[col] : 0.0f;
            float bi = (bias  && valid) ? bias[col]  : 0.0f;
#pragma unroll
            for (int r = 0; r < 8; ++r) {
                int m = m0 + r;
                float y = acc[i][j][r];
                if (scale) y = y * sc + sh;
                else       y += bi;
                if (resid) y += (float)resid[(size_t)m * ldr + col];
                if (act == 1)      y = fmaxf(y, 0.0f);
                else if (act == 2) y = 0.5f * y * (1.0f + erff(y * 0.70710678118654752f));
                if (outh)          outh[(size_t)m * ldo + col] = (_Float16)y;
                if (outf && valid) outf[(size_t)m * ldof + col] = y;
            }
        }
    }
}

// ---------------------------------------------------------------------------
// Row LayerNorm (f16 in/out, f32 math). One 256-thread block per row, C<=1024.
// Safe in-place (row cached in registers). Optional fused ReLU.
// ---------------------------------------------------------------------------
__global__ __launch_bounds__(256)
void k_layernorm(const _Float16* __restrict__ X, int ld,
                 const float* __restrict__ g, const float* __restrict__ b,
                 _Float16* __restrict__ Y, int ldo, int C, int relu) {
    __shared__ float red[256];
    int row = blockIdx.x, t = threadIdx.x;
    const _Float16* x = X + (size_t)row * ld;
    float xv[4];
    int cnt = 0;
    float s = 0.f;
    for (int c = t; c < C; c += 256) { float v = (float)x[c]; xv[cnt++] = v; s += v; }
    red[t] = s; __syncthreads();
    for (int o = 128; o > 0; o >>= 1) { if (t < o) red[t] += red[t + o]; __syncthreads(); }
    float mean = red[0] / C; __syncthreads();
    float vs = 0.f;
    for (int i = 0; i < cnt; ++i) { float d = xv[i] - mean; vs += d * d; }
    red[t] = vs; __syncthreads();
    for (int o = 128; o > 0; o >>= 1) { if (t < o) red[t] += red[t + o]; __syncthreads(); }
    float inv = rsqrtf(red[0] / C + 1e-5f);
    _Float16* y = Y + (size_t)row * ldo;
    int i = 0;
    for (int c = t; c < C; c += 256) {
        float v = (xv[i++] - mean) * inv * g[c] + b[c];
        if (relu) v = fmaxf(v, 0.f);
        y[c] = (_Float16)v;
    }
}

// ---------------------------------------------------------------------------
// Attention: one 64-thread block per (b, h, q) row. hd = 32, T = 544.
// qkv layout per token row (768 f16): [q 256 | k 256 | v 256], head h at h*32.
// ---------------------------------------------------------------------------
__global__ __launch_bounds__(64)
void k_attn(const _Float16* __restrict__ qkv, _Float16* __restrict__ out) {
    __shared__ float sc[Tt];
    __shared__ float red[64];
    int t = threadIdx.x;
    int bid = blockIdx.x;
    int q = bid % Tt;
    int h = (bid / Tt) & (HEADS - 1);
    int b = bid / (Tt * HEADS);
    const _Float16* base = qkv + (size_t)b * Tt * (3 * HID);
    const _Float16* Q = base + (size_t)q * (3 * HID) + h * HD;
    float qv[HD];
#pragma unroll
    for (int d = 0; d < HD; ++d) qv[d] = (float)Q[d] * 0.17677669529663687f; // hd^-0.5

    for (int k = t; k < Tt; k += 64) {
        const _Float16* Kp = base + (size_t)k * (3 * HID) + HID + h * HD;
        float s = 0.f;
#pragma unroll
        for (int d = 0; d < HD; ++d) s += qv[d] * (float)Kp[d];
        sc[k] = s;
    }
    __syncthreads();
    float mx = -1e30f;
    for (int k = t; k < Tt; k += 64) mx = fmaxf(mx, sc[k]);
    red[t] = mx; __syncthreads();
    for (int o = 32; o > 0; o >>= 1) { if (t < o) red[t] = fmaxf(red[t], red[t + o]); __syncthreads(); }
    mx = red[0]; __syncthreads();
    float sum = 0.f;
    for (int k = t; k < Tt; k += 64) { float e = expf(sc[k] - mx); sc[k] = e; sum += e; }
    red[t] = sum; __syncthreads();
    for (int o = 32; o > 0; o >>= 1) { if (t < o) red[t] += red[t + o]; __syncthreads(); }
    float inv = 1.0f / red[0];
    __syncthreads();
    if (t < HD) {
        float o1 = 0.f;
        for (int k = 0; k < Tt; ++k)
            o1 += sc[k] * (float)base[(size_t)k * (3 * HID) + 2 * HID + h * HD + t];
        out[((size_t)(b * Tt + q)) * HID + h * HD + t] = (_Float16)(o1 * inv);
    }
}

// ---------------------------------------------------------------------------
// Prompt: mean over first P tokens per (b, c) -> f32 [B][HID]
// ---------------------------------------------------------------------------
__global__ __launch_bounds__(256)
void k_prompt(const _Float16* __restrict__ xa, float* __restrict__ prompt) {
    int b = blockIdx.x, c = threadIdx.x;
    float s = 0.f;
    for (int t = 0; t < Pp; ++t) s += (float)xa[((size_t)(b * Tt + t)) * HID + c];
    prompt[b * HID + c] = s * (1.0f / Pp);
}

// ---------------------------------------------------------------------------
// Global-feature max over G rows -> f32 [B][EMB]
// ---------------------------------------------------------------------------
__global__ __launch_bounds__(64)
void k_gmax(const float* __restrict__ g, float* __restrict__ gmax) {
    int b = blockIdx.x, e = threadIdx.x;
    float mx = -1e30f;
    for (int r = 0; r < Gg; ++r) mx = fmaxf(mx, g[((size_t)b * Gg + r) * EMB + e]);
    gmax[b * EMB + e] = mx;
}

// ---------------------------------------------------------------------------
// Top-3 nearest centers per point + inverse-distance weights.
// ---------------------------------------------------------------------------
__global__ __launch_bounds__(256)
void k_knn(const float* __restrict__ xyz, const float* __restrict__ centers,
           int* __restrict__ idx, float* __restrict__ wgt) {
    long i = (long)blockIdx.x * 256 + threadIdx.x;
    if (i >= MP) return;
    int b = (int)(i / Nn);
    float px = xyz[i * 3 + 0], py = xyz[i * 3 + 1], pz = xyz[i * 3 + 2];
    const float* cb = centers + (size_t)b * Ss * 3;
    float d0 = 1e30f, d1 = 1e30f, d2 = 1e30f;
    int   i0 = 0, i1 = 0, i2 = 0;
    for (int s = 0; s < Ss; ++s) {
        float dx = px - cb[s * 3 + 0];
        float dy = py - cb[s * 3 + 1];
        float dz = pz - cb[s * 3 + 2];
        float d = dx * dx + dy * dy + dz * dz;
        if (d < d0)      { d2 = d1; i2 = i1; d1 = d0; i1 = i0; d0 = d; i0 = s; }
        else if (d < d1) { d2 = d1; i2 = i1; d1 = d;  i1 = s; }
        else if (d < d2) { d2 = d;  i2 = s; }
    }
    float r0 = 1.0f / (d0 + 1e-8f), r1 = 1.0f / (d1 + 1e-8f), r2 = 1.0f / (d2 + 1e-8f);
    float inv = 1.0f / (r0 + r1 + r2);
    idx[i * 3 + 0] = i0; idx[i * 3 + 1] = i1; idx[i * 3 + 2] = i2;
    wgt[i * 3 + 0] = r0 * inv; wgt[i * 3 + 1] = r1 * inv; wgt[i * 3 + 2] = r2 * inv;
}

// ---------------------------------------------------------------------------
// Build fp-stage input row [xyz(3) | w0*tok0(256) | w1*tok1 | w2*tok2 | 0-pad]
// f16, width FPK=800. One block per point.
// ---------------------------------------------------------------------------
__global__ __launch_bounds__(256)
void k_build_xp(const float* __restrict__ xyz, const _Float16* __restrict__ xa,
                const int* __restrict__ idx, const float* __restrict__ wgt,
                _Float16* __restrict__ Xp) {
    long row = blockIdx.x;
    int  b   = (int)(row / Nn);
    int  t   = threadIdx.x;
    const int*   id = idx + row * 3;
    const float* w  = wgt + row * 3;
    _Float16* dst = Xp + (size_t)row * FPK;
#pragma unroll
    for (int k = 0; k < 3; ++k) {
        const _Float16* tok = xa + ((size_t)(b * Tt + Pp + id[k])) * HID;
        dst[3 + k * HID + t] = (_Float16)(w[k] * (float)tok[t]);
    }
    if (t < 3)            dst[t] = (_Float16)xyz[row * 3 + t];
    if (t >= 3 && t < 32) dst[768 + t] = (_Float16)0.0f;   // cols 771..799
}

// ---------------------------------------------------------------------------
// Build classifier input row [h(512) | prompt(256) | gmax(64) | obj(64)] f16.
// ---------------------------------------------------------------------------
__global__ __launch_bounds__(256)
void k_build_z(const _Float16* __restrict__ h2, const float* __restrict__ prompt,
               const float* __restrict__ gmax, const float* __restrict__ obj,
               const int* __restrict__ cls, _Float16* __restrict__ z) {
    long row = blockIdx.x;
    int  b   = (int)(row / Nn);
    int  lab = cls[b];
    _Float16* dst = z + (size_t)row * CLSK;
    for (int c = threadIdx.x; c < CLSK; c += 256) {
        float v;
        if (c < 512)      v = (float)h2[(size_t)row * 512 + c];
        else if (c < 768) v = prompt[b * HID + (c - 512)];
        else if (c < 832) v = gmax[b * EMB + (c - 768)];
        else              v = obj[(size_t)lab * OBJD + (c - 832)];
        dst[c] = (_Float16)v;
    }
}

// ---------------------------------------------------------------------------
// Host orchestration
// ---------------------------------------------------------------------------
extern "C" void kernel_launch(void* const* d_in, const int* in_sizes, int n_in,
                              void* d_out, int out_size, void* d_ws, size_t ws_size,
                              hipStream_t stream) {
    (void)in_sizes; (void)n_in; (void)out_size; (void)ws_size;

    // Input mapping (insertion-order flatten of setup_inputs(), params recursed
    // in insertion order): 0..6 tensors, 7 obj_emb, 8+14*i block i leaves,
    // 50..61 fp leaves, 62..71 head leaves.
    const float* feat[3]   = {(const float*)d_in[0], (const float*)d_in[1], (const float*)d_in[2]};
    const int*   cls       = (const int*)d_in[3];
    const float* xyz       = (const float*)d_in[4];
    const float* centers   = (const float*)d_in[5];
    const float* gfeat     = (const float*)d_in[6];
    const float* obj_emb   = (const float*)d_in[7];
    auto blkp = [&](int i, int j) { return (const float*)d_in[8 + 14 * i + j]; };
    // j: 0 red_w,1 red_b,2 n1_g,3 n1_b,4 qkv_w,5 qkv_b,6 proj_w,7 proj_b,
    //    8 n2_g,9 n2_b,10 fc1_w,11 fc1_b,12 fc2_w,13 fc2_b
    auto fpp  = [&](int j) { return (const float*)d_in[50 + j]; };
    // j: 0 w1,1 b1,2 bn1_g,3 bn1_b,4 bn1_m,5 bn1_v,6 w2,7 b2,8 bn2_g,9 bn2_b,10 bn2_m,11 bn2_v
    auto hdp  = [&](int j) { return (const float*)d_in[62 + j]; };
    // j: 0 w1,1 b1,2 ln1_g,3 ln1_b,4 w2,5 b2,6 ln2_g,7 ln2_b,8 w3,9 b3

    // --- workspace arena ---------------------------------------------------
    char* base = (char*)d_ws;
    size_t off = 0;
    auto alloc = [&](size_t bytes) -> char* {
        off = (off + 255) & ~(size_t)255;
        char* p = base + off;
        off += bytes;
        return p;
    };
    auto halves = [](long n) { return (size_t)n * sizeof(_Float16); };

    char* wred[3];  char* wqkv[3]; char* wproj[3]; char* wfc1[3]; char* wfc2[3];
    for (int i = 0; i < 3; ++i) {
        wred[i]  = alloc(halves((long)INC * HID));
        wqkv[i]  = alloc(halves((long)HID * 3 * HID));
        wproj[i] = alloc(halves((long)HID * HID));
        wfc1[i]  = alloc(halves((long)HID * MLP));
        wfc2[i]  = alloc(halves((long)MLP * HID));
    }
    char* wfp1 = alloc(halves((long)FPK * 1024));
    char* wfp2 = alloc(halves((long)1024 * 512));
    char* whd1 = alloc(halves((long)CLSK * 512));
    char* whd2 = alloc(halves((long)512 * 256));
    char* whd3 = alloc(halves((long)256 * 64));

    char* feath = alloc(halves((long)MT * INC));
    char* xa    = alloc(halves((long)MT * HID));
    char* xb    = alloc(halves((long)MT * HID));
    char* xl    = alloc(halves((long)MT * HID));
    char* qkvb  = alloc(halves((long)MT * 3 * HID));
    char* ao    = alloc(halves((long)MT * HID));
    char* hb    = alloc(halves((long)MT * MLP));
    float* promptf = (float*)alloc(Bz * HID * 4);
    float* gmaxf   = (float*)alloc(Bz * EMB * 4);
    float* sc1 = (float*)alloc(1024 * 4);
    float* sh1 = (float*)alloc(1024 * 4);
    float* sc2 = (float*)alloc(512 * 4);
    float* sh2 = (float*)alloc(512 * 4);
    int*   idx = (int*)alloc(MP * 3 * 4);
    float* wgt = (float*)alloc(MP * 3 * 4);
    // Region A: Xp (MP*800 f16); later reused for h2 (MP*512) then y1 / y2.
    char* regA = alloc(halves(MP * (long)FPK));
    // Region B: h1 (MP*1024 f16); later reused for z (MP*896).
    char* regB = alloc(halves(MP * 1024L));
    char* Xp = regA;
    char* h1 = regB;
    char* h2 = regA;                              // after Xp is dead
    char* z  = regB;                              // after h1 is dead
    char* y1 = regA;                              // after h2 is dead
    char* y2 = regA + halves(MP * 512L);          // alongside y1

    // --- helpers -----------------------------------------------------------
    auto cvt = [&](const float* s, char* d, long n) {
        long blocks = (n + 255) / 256;
        k_f32_to_f16<<<dim3((unsigned)blocks), dim3(256), 0, stream>>>(s, (_Float16*)d, n);
    };
    auto pack = [&](const float* W, int K, int Nc, int Kp, int Np, char* dst) {
        long total = (long)(Kp / 32) * (Np / 16) * 512;
        long blocks = (total + 255) / 256;
        k_pack_b<<<dim3((unsigned)blocks), dim3(256), 0, stream>>>(W, K, Nc, Kp, Np,
                                                                   (_Float16*)dst, total);
    };
    auto gemm = [&](const char* A, int lda, const char* Bp, const float* bias,
                    const float* scl, const float* shf, const char* resid, int ldr,
                    char* outh, int ldo, float* outf, int ldof, int nstore,
                    int M, int N, int Kp, int act) {
        int waves = (M / 32) * (N / 64);          // one wave per 32x64 tile
        int blocks = (waves + 7) / 8;
        k_wmma_gemm<<<dim3(blocks), dim3(256), 0, stream>>>(
            (const _Float16*)A, lda, (const _Float16*)Bp, bias, scl, shf,
            (const _Float16*)resid, ldr, (_Float16*)outh, ldo,
            outf, ldof, nstore, M, N, Kp, act);
    };
    auto ln = [&](const char* X, int ld, const float* g, const float* b,
                  char* Y, int ldo, int C, int relu, long rows) {
        k_layernorm<<<dim3((unsigned)rows), dim3(256), 0, stream>>>(
            (const _Float16*)X, ld, g, b, (_Float16*)Y, ldo, C, relu);
    };

    // --- pack all weights (deterministic, every call) ----------------------
    for (int i = 0; i < 3; ++i) {
        pack(blkp(i, 0),  INC, HID,     INC, HID,     wred[i]);
        pack(blkp(i, 4),  HID, 3 * HID, HID, 3 * HID, wqkv[i]);
        pack(blkp(i, 6),  HID, HID,     HID, HID,     wproj[i]);
        pack(blkp(i, 10), HID, MLP,     HID, MLP,     wfc1[i]);
        pack(blkp(i, 12), MLP, HID,     MLP, HID,     wfc2[i]);
    }
    pack(fpp(0), 771, 1024, FPK, 1024, wfp1);
    pack(fpp(6), 1024, 512, 1024, 512, wfp2);
    pack(hdp(0), CLSK, 512, CLSK, 512, whd1);
    pack(hdp(4), 512, 256, 512, 256, whd2);
    pack(hdp(8), 256, OUTC, 256, 64, whd3);
    k_bn_fold<<<dim3(4), dim3(256), 0, stream>>>(fpp(1), fpp(2), fpp(3), fpp(4), fpp(5), sc1, sh1, 1024);
    k_bn_fold<<<dim3(2), dim3(256), 0, stream>>>(fpp(7), fpp(8), fpp(9), fpp(10), fpp(11), sc2, sh2, 512);

    // --- transformer backbone: blocks use feat2, feat1, feat0 --------------
    for (int i = 0; i < 3; ++i) {
        cvt(feat[2 - i], feath, (long)MT * INC);
        // x = feat @ red_w + red_b (+ x)
        gemm(feath, INC, wred[i], blkp(i, 1), nullptr, nullptr,
             i ? xa : nullptr, HID, xa, HID, nullptr, 0, HID, MT, HID, INC, 0);
        // attn branch
        ln(xa, HID, blkp(i, 2), blkp(i, 3), xl, HID, HID, 0, MT);
        gemm(xl, HID, wqkv[i], blkp(i, 5), nullptr, nullptr,
             nullptr, 0, qkvb, 3 * HID, nullptr, 0, 3 * HID, MT, 3 * HID, HID, 0);
        k_attn<<<dim3(Bz * HEADS * Tt), dim3(64), 0, stream>>>((const _Float16*)qkvb,
                                                               (_Float16*)ao);
        gemm(ao, HID, wproj[i], blkp(i, 7), nullptr, nullptr,
             xa, HID, xb, HID, nullptr, 0, HID, MT, HID, HID, 0);
        // mlp branch
        ln(xb, HID, blkp(i, 8), blkp(i, 9), xl, HID, HID, 0, MT);
        gemm(xl, HID, wfc1[i], blkp(i, 11), nullptr, nullptr,
             nullptr, 0, hb, MLP, nullptr, 0, MLP, MT, MLP, HID, 2 /*gelu*/);
        gemm(hb, MLP, wfc2[i], blkp(i, 13), nullptr, nullptr,
             xb, HID, xa, HID, nullptr, 0, HID, MT, HID, MLP, 0);
    }

    // --- feature propagation inputs ---------------------------------------
    k_prompt<<<dim3(Bz), dim3(HID), 0, stream>>>((const _Float16*)xa, promptf);
    k_knn<<<dim3((unsigned)(MP / 256)), dim3(256), 0, stream>>>(xyz, centers, idx, wgt);
    k_build_xp<<<dim3((unsigned)MP), dim3(256), 0, stream>>>(xyz, (const _Float16*)xa,
                                                             idx, wgt, (_Float16*)Xp);

    // --- per-point MLPs (the FLOP-dominant stage) --------------------------
    gemm(Xp, FPK, wfp1, nullptr, sc1, sh1, nullptr, 0,
         h1, 1024, nullptr, 0, 1024, (int)MP, 1024, FPK, 1 /*relu*/);
    gemm(h1, 1024, wfp2, nullptr, sc2, sh2, nullptr, 0,
         h2, 512, nullptr, 0, 512, (int)MP, 512, 1024, 1 /*relu*/);

    k_gmax<<<dim3(Bz), dim3(EMB), 0, stream>>>(gfeat, gmaxf);
    k_build_z<<<dim3((unsigned)MP), dim3(256), 0, stream>>>((const _Float16*)h2, promptf,
                                                            gmaxf, obj_emb, cls, (_Float16*)z);

    // --- classification head ----------------------------------------------
    gemm(z, CLSK, whd1, hdp(1), nullptr, nullptr, nullptr, 0,
         y1, 512, nullptr, 0, 512, (int)MP, 512, CLSK, 0);
    ln(y1, 512, hdp(2), hdp(3), y1, 512, 512, 1 /*relu*/, MP);
    gemm(y1, 512, whd2, hdp(5), nullptr, nullptr, nullptr, 0,
         y2, 256, nullptr, 0, 256, (int)MP, 256, 512, 0);
    ln(y2, 256, hdp(6), hdp(7), y2, 256, 256, 1 /*relu*/, MP);
    gemm(y2, 256, whd3, hdp(9), nullptr, nullptr, nullptr, 0,
         nullptr, 0, (float*)d_out, OUTC, OUTC, (int)MP, 64, 256, 0);
}
